// MemoryAttention_80075370266649
// MI455X (gfx1250) — matrix operand order repaired and checked by reading, not compiled, not run on hardware
//
#include <hip/hip_runtime.h>
#include <hip/hip_bf16.h>

#define S_  512
#define B_  8
#define C_  1024
#define H_  16
#define HC_ 64

typedef __attribute__((ext_vector_type(16))) __bf16         v16bf;
typedef __attribute__((ext_vector_type(8)))  float          v8f;
typedef __attribute__((ext_vector_type(8)))  unsigned short ush8;
typedef unsigned short u16t;

union FragBF {
    ush8  h[2];
    v16bf v;
};
union Pack8 {
    u16t u[8];
    ush8 v;
};

// native bf16 convert: clang lowers fptrunc f32->bf16 to HW cvt on gfx1250
__device__ __forceinline__ u16t f2bf(float f) {
    return __builtin_bit_cast(u16t, (__bf16)f);
}
__device__ __forceinline__ float bf2f(u16t h) {
    return (float)__builtin_bit_cast(__bf16, h);
}

// ---------------------------------------------------------------------------
// Kernel 1: qkv = x @ Wqkv + bqkv ; scatter to q/k/v bf16 buffers (b,h,s,c)
// Workgroup: 256 threads (8 waves). Tile: 128 rows x 64 cols, K-step 32.
// ---------------------------------------------------------------------------
__global__ __launch_bounds__(256)
void qkv_gemm_kernel(const float* __restrict__ x,
                     const float* __restrict__ Wqkv,
                     const float* __restrict__ bqkv,
                     u16t* __restrict__ qbf,
                     u16t* __restrict__ kbf,
                     u16t* __restrict__ vbf)
{
    __shared__ u16t abuf[128 * 32];
    __shared__ u16t bbuf[64 * 32];

    const int tid    = threadIdx.x;
    const int lane   = tid & 31;
    const int wave   = tid >> 5;
    const int laneLo = lane & 15;
    const int hiH    = lane >> 4;
    const int Mbase  = blockIdx.y * 128;
    const int Nbase  = blockIdx.x * 64;

    const int arow  = tid >> 1;
    const int akoff = (tid & 1) * 16;
    const int bn    = tid & 63;
    const int bkb   = (tid >> 6) * 8;

    v8f acc[4];
    #pragma unroll
    for (int i = 0; i < 4; ++i) acc[i] = {};

    for (int k0 = 0; k0 < C_; k0 += 32) {
        // stage A: 128x32 fp32 -> bf16 (row-major, 32 per row)
        {
            const float* srcA = x + (size_t)(Mbase + arow) * C_ + k0 + akoff;
            if (k0 + 32 < C_) __builtin_prefetch(srcA + 32, 0, 0);   // next K tile
            Pack8 p0, p1;
            #pragma unroll
            for (int i = 0; i < 2; ++i) {
                float4 v = reinterpret_cast<const float4*>(srcA)[i];
                p0.u[i * 4 + 0] = f2bf(v.x); p0.u[i * 4 + 1] = f2bf(v.y);
                p0.u[i * 4 + 2] = f2bf(v.z); p0.u[i * 4 + 3] = f2bf(v.w);
            }
            #pragma unroll
            for (int i = 0; i < 2; ++i) {
                float4 v = reinterpret_cast<const float4*>(srcA)[2 + i];
                p1.u[i * 4 + 0] = f2bf(v.x); p1.u[i * 4 + 1] = f2bf(v.y);
                p1.u[i * 4 + 2] = f2bf(v.z); p1.u[i * 4 + 3] = f2bf(v.w);
            }
            *reinterpret_cast<ush8*>(&abuf[arow * 32 + akoff])     = p0.v;
            *reinterpret_cast<ush8*>(&abuf[arow * 32 + akoff + 8]) = p1.v;
        }
        // stage B: 32x64 of Wqkv, stored column(K)-major in LDS
        {
            const float* srcB = Wqkv + (size_t)(k0 + bkb) * (3 * C_) + Nbase + bn;
            if (k0 + 32 < C_) __builtin_prefetch(srcB + 32 * (3 * C_), 0, 0);
            Pack8 p;
            #pragma unroll
            for (int i = 0; i < 8; ++i)
                p.u[i] = f2bf(srcB[(size_t)i * (3 * C_)]);
            *reinterpret_cast<ush8*>(&bbuf[bn * 32 + bkb]) = p.v;
        }
        __syncthreads();

        FragBF a;
        a.h[0] = *reinterpret_cast<const ush8*>(&abuf[(wave * 16 + laneLo) * 32 + hiH * 8]);
        a.h[1] = *reinterpret_cast<const ush8*>(&abuf[(wave * 16 + laneLo) * 32 + 16 + hiH * 8]);
        #pragma unroll
        for (int ct = 0; ct < 4; ++ct) {
            FragBF bfr;
            bfr.h[0] = *reinterpret_cast<const ush8*>(&bbuf[(ct * 16 + laneLo) * 32 + hiH * 8]);
            bfr.h[1] = *reinterpret_cast<const ush8*>(&bbuf[(ct * 16 + laneLo) * 32 + 16 + hiH * 8]);
            acc[ct] = __builtin_amdgcn_wmma_f32_16x16x32_bf16(
                false, a.v, false, bfr.v, (short)0, acc[ct], false, false);
        }
        __syncthreads();
    }

    // epilogue: +bias, scatter into q/k/v (b,h,s,c) bf16
    #pragma unroll
    for (int ct = 0; ct < 4; ++ct) {
        const int n    = Nbase + ct * 16 + laneLo;
        const int part = n >> 10;
        const int rem  = n & 1023;
        const int h    = rem >> 6;
        const int c    = rem & 63;
        const float bias = bqkv[n];
        u16t* dst = (part == 0) ? qbf : (part == 1) ? kbf : vbf;
        #pragma unroll
        for (int j = 0; j < 8; ++j) {
            const int row = Mbase + wave * 16 + j + hiH * 8;
            const int s = row >> 3, b = row & 7;
            dst[(((size_t)b * H_ + h) * S_ + s) * HC_ + c] = f2bf(acc[ct][j] + bias);
        }
    }
}

// ---------------------------------------------------------------------------
// Kernel 2: flash attention per (b,h) and 64-query block.
// 128 threads = 4 waves x 16 query rows. Valid keys = prefix [0, vlen).
// ---------------------------------------------------------------------------
__global__ __launch_bounds__(128)
void attn_kernel(const u16t* __restrict__ qbf,
                 const u16t* __restrict__ kbf,
                 const u16t* __restrict__ vbf,
                 const float* __restrict__ memory_kv,
                 const int* __restrict__ memory_length,
                 const int* __restrict__ pad_q,
                 u16t* __restrict__ obf)
{
    __shared__ u16t ksm[32 * 64];      // [key][c]  key-major
    __shared__ u16t vsm[64 * 32];      // [c][key]  channel-major (transposed)
    __shared__ u16t psm[4][16 * 32];   // per-wave P tile for D->A transpose

    const int tid    = threadIdx.x;
    const int lane   = tid & 31;
    const int wave   = tid >> 5;
    const int laneLo = lane & 15;
    const int hiH    = lane >> 4;
    const int bh     = blockIdx.x;         // b*16 + h
    const int b      = bh >> 4;
    const int h      = bh & 15;
    const int qbase  = blockIdx.y * 64 + wave * 16;

    const int L  = memory_length[b];
    int Pq = pad_q[b]; if (Pq > S_) Pq = S_;
    const int vlen = L + Pq;               // mask is a pure prefix length

    // Q fragments (rows qbase..qbase+15, K=64 -> two 32-chunks)
    const u16t* qrow = qbf + (((size_t)b * H_ + h) * S_ + qbase + laneLo) * HC_;
    FragBF qf[2];
    #pragma unroll
    for (int kk = 0; kk < 2; ++kk) {
        qf[kk].h[0] = *reinterpret_cast<const ush8*>(qrow + kk * 32 + hiH * 8);
        qf[kk].h[1] = *reinterpret_cast<const ush8*>(qrow + kk * 32 + 16 + hiH * 8);
    }

    float mrow[8], lrow[8];
    v8f oacc[4];
    #pragma unroll
    for (int j = 0; j < 8; ++j) { mrow[j] = -3.0e38f; lrow[j] = 0.0f; }
    #pragma unroll
    for (int ct = 0; ct < 4; ++ct) oacc[ct] = {};

    const u16t* kbhd = kbf + ((size_t)b * H_ + h) * S_ * HC_;
    const u16t* vbhd = vbf + ((size_t)b * H_ + h) * S_ * HC_;

    const int kl = tid >> 2;           // key local 0..31 (staging role)
    const int cg = (tid & 3) * 16;     // channel group   (staging role)

    for (int t0 = 0; t0 < vlen; t0 += 32) {
        // ---- cooperative staging of 32 keys x 64 channels of K and V ----
        {
            const int t = t0 + kl;
            u16t karr[16], varr[16];
            if (t < L) {
                const float* src = memory_kv + (((size_t)t * B_ + b) * H_ + h) * (2 * HC_);
                if (t + 32 < L)
                    __builtin_prefetch(src + (size_t)32 * B_ * H_ * 2 * HC_ + cg, 0, 0);
                #pragma unroll
                for (int i = 0; i < 4; ++i) {
                    float4 vk = *reinterpret_cast<const float4*>(src + cg + i * 4);
                    float4 vv = *reinterpret_cast<const float4*>(src + HC_ + cg + i * 4);
                    karr[i * 4 + 0] = f2bf(vk.x); karr[i * 4 + 1] = f2bf(vk.y);
                    karr[i * 4 + 2] = f2bf(vk.z); karr[i * 4 + 3] = f2bf(vk.w);
                    varr[i * 4 + 0] = f2bf(vv.x); varr[i * 4 + 1] = f2bf(vv.y);
                    varr[i * 4 + 2] = f2bf(vv.z); varr[i * 4 + 3] = f2bf(vv.w);
                }
            } else if (t - L < S_) {
                const u16t* sk = kbhd + (size_t)(t - L) * HC_ + cg;
                const u16t* sv = vbhd + (size_t)(t - L) * HC_ + cg;
                Pack8 k0p, k1p, v0p, v1p;
                k0p.v = reinterpret_cast<const ush8*>(sk)[0];
                k1p.v = reinterpret_cast<const ush8*>(sk)[1];
                v0p.v = reinterpret_cast<const ush8*>(sv)[0];
                v1p.v = reinterpret_cast<const ush8*>(sv)[1];
                #pragma unroll
                for (int i = 0; i < 8; ++i) {
                    karr[i] = k0p.u[i]; karr[8 + i] = k1p.u[i];
                    varr[i] = v0p.u[i]; varr[8 + i] = v1p.u[i];
                }
            } else {
                #pragma unroll
                for (int i = 0; i < 16; ++i) { karr[i] = 0; varr[i] = 0; }
            }
            Pack8 p0, p1;
            #pragma unroll
            for (int i = 0; i < 8; ++i) { p0.u[i] = karr[i]; p1.u[i] = karr[8 + i]; }
            *reinterpret_cast<ush8*>(&ksm[kl * 64 + cg])     = p0.v;   // ds_store_b128
            *reinterpret_cast<ush8*>(&ksm[kl * 64 + cg + 8]) = p1.v;
            #pragma unroll
            for (int i = 0; i < 16; ++i)                              // transposed scatter
                vsm[(cg + i) * 32 + kl] = varr[i];
        }
        __syncthreads();

        // ---- QK^T: two 16-key score tiles, K=64 in two 32-chunks ----
        float sc[2][8];
        #pragma unroll
        for (int kh = 0; kh < 2; ++kh) {
            v8f s = {};
            #pragma unroll
            for (int kk = 0; kk < 2; ++kk) {
                FragBF kfr;
                const int key = kh * 16 + laneLo;
                kfr.h[0] = *reinterpret_cast<const ush8*>(&ksm[key * 64 + kk * 32 + hiH * 8]);
                kfr.h[1] = *reinterpret_cast<const ush8*>(&ksm[key * 64 + kk * 32 + 16 + hiH * 8]);
                s = __builtin_amdgcn_wmma_f32_16x16x32_bf16(
                    false, qf[kk].v, false, kfr.v, (short)0, s, false, false);
            }
            const bool valid = (t0 + kh * 16 + laneLo) < vlen;
            #pragma unroll
            for (int j = 0; j < 8; ++j)
                sc[kh][j] = valid ? s[j] * 0.125f : -3.0e38f;   // 1/sqrt(64)
        }

        // ---- online softmax: row stats live per lane-half (matches C/D layout) ----
        float cmax[8];
        #pragma unroll
        for (int j = 0; j < 8; ++j) cmax[j] = fmaxf(sc[0][j], sc[1][j]);
        #pragma unroll
        for (int d = 1; d <= 8; d <<= 1) {
            #pragma unroll
            for (int j = 0; j < 8; ++j)
                cmax[j] = fmaxf(cmax[j], __shfl_xor(cmax[j], d, 32));
        }
        float alpha[8], rsum[8];
        #pragma unroll
        for (int j = 0; j < 8; ++j) {
            float nm = fmaxf(mrow[j], cmax[j]);
            alpha[j] = __expf(mrow[j] - nm);
            mrow[j]  = nm;
            sc[0][j] = __expf(sc[0][j] - nm);
            sc[1][j] = __expf(sc[1][j] - nm);
            rsum[j]  = sc[0][j] + sc[1][j];
        }
        #pragma unroll
        for (int d = 1; d <= 8; d <<= 1) {
            #pragma unroll
            for (int j = 0; j < 8; ++j)
                rsum[j] += __shfl_xor(rsum[j], d, 32);
        }
        #pragma unroll
        for (int j = 0; j < 8; ++j) lrow[j] = lrow[j] * alpha[j] + rsum[j];
        #pragma unroll
        for (int ct = 0; ct < 4; ++ct)
            #pragma unroll
            for (int j = 0; j < 8; ++j) oacc[ct][j] *= alpha[j];

        // ---- P (D layout) -> LDS -> A fragment (per-wave region, DS in-order) ----
        #pragma unroll
        for (int kh = 0; kh < 2; ++kh)
            #pragma unroll
            for (int j = 0; j < 8; ++j) {
                const int row = j + hiH * 8;
                psm[wave][row * 32 + kh * 16 + laneLo] = f2bf(sc[kh][j]);
            }
        FragBF pf;
        pf.h[0] = *reinterpret_cast<const ush8*>(&psm[wave][laneLo * 32 + hiH * 8]);
        pf.h[1] = *reinterpret_cast<const ush8*>(&psm[wave][laneLo * 32 + 16 + hiH * 8]);

        // ---- P @ V ----
        #pragma unroll
        for (int ct = 0; ct < 4; ++ct) {
            FragBF vf;
            vf.h[0] = *reinterpret_cast<const ush8*>(&vsm[(ct * 16 + laneLo) * 32 + hiH * 8]);
            vf.h[1] = *reinterpret_cast<const ush8*>(&vsm[(ct * 16 + laneLo) * 32 + 16 + hiH * 8]);
            oacc[ct] = __builtin_amdgcn_wmma_f32_16x16x32_bf16(
                false, pf.v, false, vf.v, (short)0, oacc[ct], false, false);
        }
        __syncthreads();
    }

    // ---- finalize: O/l -> bf16 attn output, row (s*B+b), col (h*64+c) ----
    float inv[8];
    #pragma unroll
    for (int j = 0; j < 8; ++j) inv[j] = (lrow[j] > 0.0f) ? 1.0f / lrow[j] : 0.0f;
    #pragma unroll
    for (int ct = 0; ct < 4; ++ct) {
        const int c = h * HC_ + ct * 16 + laneLo;
        #pragma unroll
        for (int j = 0; j < 8; ++j) {
            const int s = qbase + j + hiH * 8;
            obf[((size_t)s * B_ + b) * C_ + c] = f2bf(oacc[ct][j] * inv[j]);
        }
    }
}

// ---------------------------------------------------------------------------
// Kernel 3: out = attn_out(bf16) @ Wc + bc  -> fp32 d_out (S,B,C)
// ---------------------------------------------------------------------------
__global__ __launch_bounds__(256)
void out_gemm_kernel(const u16t* __restrict__ abf,
                     const float* __restrict__ Wc,
                     const float* __restrict__ bc,
                     float* __restrict__ out)
{
    __shared__ u16t abuf[128 * 32];
    __shared__ u16t bbuf[64 * 32];

    const int tid    = threadIdx.x;
    const int lane   = tid & 31;
    const int wave   = tid >> 5;
    const int laneLo = lane & 15;
    const int hiH    = lane >> 4;
    const int Mbase  = blockIdx.y * 128;
    const int Nbase  = blockIdx.x * 64;

    const int arow  = tid >> 1;
    const int akoff = (tid & 1) * 16;
    const int bn    = tid & 63;
    const int bkb   = (tid >> 6) * 8;

    v8f acc[4];
    #pragma unroll
    for (int i = 0; i < 4; ++i) acc[i] = {};

    for (int k0 = 0; k0 < C_; k0 += 32) {
        {
            const u16t* srcA = abf + (size_t)(Mbase + arow) * C_ + k0 + akoff;
            if (k0 + 32 < C_) __builtin_prefetch(srcA + 32, 0, 0);
            *reinterpret_cast<ush8*>(&abuf[arow * 32 + akoff]) =
                reinterpret_cast<const ush8*>(srcA)[0];
            *reinterpret_cast<ush8*>(&abuf[arow * 32 + akoff + 8]) =
                reinterpret_cast<const ush8*>(srcA)[1];
        }
        {
            const float* srcB = Wc + (size_t)(k0 + bkb) * C_ + Nbase + bn;
            if (k0 + 32 < C_) __builtin_prefetch(srcB + (size_t)32 * C_, 0, 0);
            Pack8 p;
            #pragma unroll
            for (int i = 0; i < 8; ++i)
                p.u[i] = f2bf(srcB[(size_t)i * C_]);
            *reinterpret_cast<ush8*>(&bbuf[bn * 32 + bkb]) = p.v;
        }
        __syncthreads();

        FragBF a;
        a.h[0] = *reinterpret_cast<const ush8*>(&abuf[(wave * 16 + laneLo) * 32 + hiH * 8]);
        a.h[1] = *reinterpret_cast<const ush8*>(&abuf[(wave * 16 + laneLo) * 32 + 16 + hiH * 8]);
        #pragma unroll
        for (int ct = 0; ct < 4; ++ct) {
            FragBF bfr;
            bfr.h[0] = *reinterpret_cast<const ush8*>(&bbuf[(ct * 16 + laneLo) * 32 + hiH * 8]);
            bfr.h[1] = *reinterpret_cast<const ush8*>(&bbuf[(ct * 16 + laneLo) * 32 + 16 + hiH * 8]);
            acc[ct] = __builtin_amdgcn_wmma_f32_16x16x32_bf16(
                false, a.v, false, bfr.v, (short)0, acc[ct], false, false);
        }
        __syncthreads();
    }

    #pragma unroll
    for (int ct = 0; ct < 4; ++ct) {
        const int n = Nbase + ct * 16 + laneLo;
        const float bias = bc[n];
        #pragma unroll
        for (int j = 0; j < 8; ++j) {
            const int row = Mbase + wave * 16 + j + hiH * 8;
            out[(size_t)row * C_ + n] = acc[ct][j] + bias;
        }
    }
}

// ---------------------------------------------------------------------------
extern "C" void kernel_launch(void* const* d_in, const int* in_sizes, int n_in,
                              void* d_out, int out_size, void* d_ws, size_t ws_size,
                              hipStream_t stream) {
    (void)in_sizes; (void)n_in; (void)out_size; (void)ws_size;

    const float* x             = (const float*)d_in[0];
    const int*   pad_q         = (const int*)d_in[1];
    /* d_in[2] = mask : unused (mask reduces to prefix length from pad_q) */
    const float* memory_kv     = (const float*)d_in[3];
    const int*   memory_length = (const int*)d_in[4];
    const float* Wqkv          = (const float*)d_in[5];
    const float* bqkv          = (const float*)d_in[6];
    const float* Wc            = (const float*)d_in[7];
    const float* bc            = (const float*)d_in[8];

    const size_t NELT = (size_t)S_ * B_ * C_;   // 4,194,304
    u16t* qbf = (u16t*)d_ws;
    u16t* kbf = qbf + NELT;
    u16t* vbf = kbf + NELT;
    u16t* obf = vbf + NELT;

    dim3 g1(3 * C_ / 64, (S_ * B_) / 128);   // 48 x 32
    qkv_gemm_kernel<<<g1, 256, 0, stream>>>(x, Wqkv, bqkv, qbf, kbf, vbf);

    dim3 g2(B_ * H_, S_ / 64);               // 128 x 8
    attn_kernel<<<g2, 128, 0, stream>>>(qbf, kbf, vbf, memory_kv, memory_length,
                                        pad_q, obf);

    dim3 g3(C_ / 64, (S_ * B_) / 128);       // 16 x 32
    out_gemm_kernel<<<g3, 256, 0, stream>>>(obf, Wc, bc, (float*)d_out);
}
